// HybridQLSTM_65481071397992
// MI455X (gfx1250) — matrix operand order repaired and checked by reading, not compiled, not run on hardware
//
#include <hip/hip_runtime.h>

// Problem constants (match reference): T=1024, B=64, D=512, H=512, O=1
#define T_STEPS 1024
#define BATCH   64
#define DIM     512
#define HID     512
#define NWG     64        // 2 batch tiles x 32 column tiles
#define XSTR    520       // padded row stride (bf16 elems) for 32x512 LDS tiles

typedef __bf16 bf16_t;
typedef __attribute__((ext_vector_type(16))) __bf16 v16bf;
typedef __attribute__((ext_vector_type(8)))  __bf16 v8bf;
typedef __attribute__((ext_vector_type(8)))  float  v8f;
typedef __attribute__((ext_vector_type(4)))  int    v4i;

union AFrag { v16bf v; v8bf h[2]; };

// CDNA5 async global->LDS copy path (guarded so the build never regresses)
#if defined(__has_builtin)
#if __has_builtin(__builtin_amdgcn_global_load_async_to_lds_b128) && \
    __has_builtin(__builtin_amdgcn_s_wait_asynccnt)
#define HAVE_ASYNC_LDS 1
#endif
#endif
#ifdef HAVE_ASYNC_LDS
#define AS1 __attribute__((address_space(1)))
#define AS3 __attribute__((address_space(3)))
#endif

__device__ __forceinline__ float sigmoid_f(float x) {
  return 1.0f / (1.0f + __expf(-x));
}
// branch-free tanh on fast exp; saturates correctly at +/-inf
__device__ __forceinline__ float tanh_f(float x) {
  return 1.0f - 2.0f / (__expf(2.0f * x) + 1.0f);
}

// stage one 32x512 fp32 tile as bf16 into a padded LDS tile (coalesced)
__device__ __forceinline__ void fill_x(bf16_t* __restrict__ dst,
                                       const float* __restrict__ src, int tid) {
  for (int idx = tid; idx < 32 * DIM; idx += 256) {
    const int d = idx & (DIM - 1);
    const int r = idx >> 9;
    dst[r * XSTR + d] = (bf16_t)src[r * DIM + d];
  }
}

// Persistent LSTM: weights LDS-resident for all 1024 steps; monotonic-counter
// grid barrier carries the recurrence; x_{t+1} is staged in the barrier shadow.
__global__ __launch_bounds__(256, 1)
void lstm_persistent(const float* __restrict__ xin,
                     const float* __restrict__ Wf, const float* __restrict__ bfv,
                     const float* __restrict__ Wi, const float* __restrict__ biv,
                     const float* __restrict__ Wg, const float* __restrict__ bgv,
                     const float* __restrict__ Wo, const float* __restrict__ bov,
                     float* __restrict__ stacked,     // [T,B,H] fp32 outputs
                     float* __restrict__ cx,          // [B,H] final cell state
                     bf16_t* __restrict__ hbuf_g,     // [2][B,H] bf16 recurrent scratch
                     unsigned* __restrict__ sync_counter)
{
  // 32 K-chunks x 4 gates x 32 lanes x 16 bf16 = 128 KB, pre-swizzled into the
  // exact WMMA B-fragment layout (per-lane contiguous -> two ds_load_b128).
  __shared__ __align__(16) bf16_t Wpack[32 * 4 * 32 * 16];
  // double-buffered x tile (32 x 512 bf16, padded rows): 2 x 32.5 KB
  __shared__ __align__(16) bf16_t xbuf[2 * 32 * XSTR];
  // h tile (32 x 512 bf16, padded rows): 32.5 KB; gate-exchange overlays it
  __shared__ __align__(16) bf16_t hlds[32 * XSTR];
  float* gatebuf = (float*)&hlds[0];       // 4 gates x 32 rows x 16 cols f32 (8 KB)

  const int tid  = threadIdx.x;
  const int lane = tid & 31;
  const int wave = tid >> 5;
  const int wg   = blockIdx.x;
  const int wgM  = wg & 1;                 // batch tile: rows [wgM*32, wgM*32+32)
  const int wgN  = wg >> 1;                // H-column tile: cols [wgN*16, ...+16)
  const int colBase = wgN * 16;

  // ---------- one-time: pack weight slice into LDS (WMMA B-frag layout) ------
  for (int idx = tid; idx < 32 * 4 * 32 * 16; idx += 256) {
    const int e   = idx & 15;
    const int ln  = (idx >> 4) & 31;
    const int g   = (idx >> 9) & 3;
    const int kc  = idx >> 11;
    const int k   = kc * 32 + ((ln >= 16) ? 8 : 0) + ((e >= 8) ? 16 : 0) + (e & 7);
    const int col = colBase + (ln & 15);
    const float* wp = (g == 0) ? Wf : (g == 1) ? Wi : (g == 2) ? Wg : Wo;
    Wpack[idx] = (bf16_t)wp[k * HID + col];
  }
  // prologue: stage x_0
  fill_x(&xbuf[0], xin + (size_t)wgM * 32 * DIM, tid);
  __syncthreads();

  // per-wave tile assignment: wave -> (batch subtile mt, gate gw)
  const int mt     = wave >> 2;
  const int gw     = wave & 3;
  const int aOff   = (mt * 16 + (lane & 15)) * XSTR + ((lane >= 16) ? 8 : 0);
  const int cn     = lane & 15;
  const int crb    = mt * 16 + ((lane >= 16) ? 8 : 0);

  // elementwise ownership: 512 cell elems per WG -> 2 per thread
  const int r0  = tid >> 4;
  const int r1  = r0 + 16;
  const int cl0 = tid & 15;
  float c0 = 0.0f, c1 = 0.0f;

  const float bF = bfv[colBase + cl0];
  const float bI = biv[colBase + cl0];
  const float bG = bgv[colBase + cl0];
  const float bO = bov[colBase + cl0];

  for (int t = 0; t < T_STEPS; ++t) {
    const bf16_t* xcur = &xbuf[(t & 1) * (32 * XSTR)];

    // ---- stage h_{t-1} (flat bf16 copy; async global->LDS when available) --
    if (t > 0) {
      bf16_t* hsrc = hbuf_g + (size_t)((t + 1) & 1) * (BATCH * HID)
                   + (size_t)wgM * 32 * HID;        // h_{t-1}, rows wgM*32..+31
#pragma unroll
      for (int k = 0; k < 8; ++k) {                  // 2048 x 16B chunks total
        const int i  = tid + k * 256;
        const int r  = i >> 6;                       // 64 chunks per 1 KB row
        const int cb = (i & 63) * 16;                // byte offset within row
#ifdef HAVE_ASYNC_LDS
        __builtin_amdgcn_global_load_async_to_lds_b128(
            (AS1 v4i*)(hsrc + (size_t)i * 8),
            (AS3 v4i*)((char*)hlds + r * (XSTR * 2) + cb), 0, 0);
#else
        *(v8bf*)((char*)hlds + r * (XSTR * 2) + cb) =
            *(const v8bf*)(hsrc + (size_t)i * 8);
#endif
      }
#ifdef HAVE_ASYNC_LDS
      __builtin_amdgcn_s_wait_asynccnt(0);
#endif
    }
    __syncthreads();

    // ---- GEMM: one 16x16 f32 tile per wave over K=1024 (h half skipped t=0)
    v8f acc = {};
#pragma unroll 4
    for (int kc = 0; kc < 16; ++kc) {                // x half: K = 0..511
      AFrag a, b;
      const bf16_t* ap = xcur + aOff + kc * 32;
      a.h[0] = *(const v8bf*)(ap);
      a.h[1] = *(const v8bf*)(ap + 16);
      const bf16_t* bp = &Wpack[(((kc << 2) | gw) * 32 + lane) << 4];
      b.h[0] = *(const v8bf*)(bp);
      b.h[1] = *(const v8bf*)(bp + 8);
      acc = __builtin_amdgcn_wmma_f32_16x16x32_bf16(false, a.v, false, b.v,
                                                    (short)0, acc, false, false);
    }
    if (t > 0) {
#pragma unroll 4
      for (int kc = 0; kc < 16; ++kc) {              // h half: K = 512..1023
        AFrag a, b;
        const bf16_t* ap = &hlds[aOff + kc * 32];
        a.h[0] = *(const v8bf*)(ap);
        a.h[1] = *(const v8bf*)(ap + 16);
        const bf16_t* bp = &Wpack[((((16 + kc) << 2) | gw) * 32 + lane) << 4];
        b.h[0] = *(const v8bf*)(bp);
        b.h[1] = *(const v8bf*)(bp + 8);
        acc = __builtin_amdgcn_wmma_f32_16x16x32_bf16(false, a.v, false, b.v,
                                                      (short)0, acc, false, false);
      }
    }
    __syncthreads();                                 // h tile consumed

    // ---- gate exchange through LDS overlay ----
#pragma unroll
    for (int r = 0; r < 8; ++r)
      gatebuf[(gw * 32 + crb + r) * 16 + cn] = acc[r];
    __syncthreads();

    // ---- LSTM cell update (2 elements/thread) ----
    bf16_t* hdst = hbuf_g + (size_t)(t & 1) * (BATCH * HID);
    {
      const float fv = sigmoid_f(gatebuf[(0 * 32 + r0) * 16 + cl0] + bF);
      const float iv = sigmoid_f(gatebuf[(1 * 32 + r0) * 16 + cl0] + bI);
      const float gv = tanh_f   (gatebuf[(2 * 32 + r0) * 16 + cl0] + bG);
      const float ov = sigmoid_f(gatebuf[(3 * 32 + r0) * 16 + cl0] + bO);
      c0 = fv * c0 + iv * gv;
      const float hv = ov * tanh_f(c0);
      stacked[((size_t)t * BATCH + wgM * 32 + r0) * HID + colBase + cl0] = hv;
      hdst[(wgM * 32 + r0) * HID + colBase + cl0] = (bf16_t)hv;
    }
    {
      const float fv = sigmoid_f(gatebuf[(0 * 32 + r1) * 16 + cl0] + bF);
      const float iv = sigmoid_f(gatebuf[(1 * 32 + r1) * 16 + cl0] + bI);
      const float gv = tanh_f   (gatebuf[(2 * 32 + r1) * 16 + cl0] + bG);
      const float ov = sigmoid_f(gatebuf[(3 * 32 + r1) * 16 + cl0] + bO);
      c1 = fv * c1 + iv * gv;
      const float hv = ov * tanh_f(c1);
      stacked[((size_t)t * BATCH + wgM * 32 + r1) * HID + colBase + cl0] = hv;
      hdst[(wgM * 32 + r1) * HID + colBase + cl0] = (bf16_t)hv;
    }

    // ---- arrive; stage x_{t+1} in the barrier shadow; then wait ----
    __threadfence();
    __syncthreads();
    if (tid == 0) atomicAdd(sync_counter, 1u);
    if (t + 1 < T_STEPS)
      fill_x(&xbuf[((t + 1) & 1) * (32 * XSTR)],
             xin + ((size_t)(t + 1) * BATCH + wgM * 32) * DIM, tid);
    if (tid == 0) {
      const unsigned want = (unsigned)(t + 1) * NWG;
      while (*(volatile unsigned*)sync_counter < want)
        __builtin_amdgcn_s_sleep(1);
    }
    __syncthreads();
    __threadfence();
  }

  // final cell state
  cx[(wgM * 32 + r0) * HID + colBase + cl0] = c0;
  cx[(wgM * 32 + r1) * HID + colBase + cl0] = c1;
}

// regress_out[t,b] = stacked[t,b,:] . Wr[:,0] + br   (O == 1)
__global__ void regress_kernel(const float* __restrict__ stacked,
                               const float* __restrict__ Wr,
                               const float* __restrict__ br,
                               float* __restrict__ out)
{
  const int row  = blockIdx.x * 8 + (threadIdx.x >> 5);   // one wave per (t,b)
  const int lane = threadIdx.x & 31;
  const float* s = stacked + (size_t)row * HID;
  float sum = 0.0f;
#pragma unroll 4
  for (int j = lane; j < HID; j += 32)
    sum += s[j] * Wr[j];
  for (int off = 16; off > 0; off >>= 1)
    sum += __shfl_down(sum, off, 32);
  if (lane == 0)
    out[row] = sum + br[0];
}

// hx = stacked[T-1]
__global__ void hx_copy_kernel(const float* __restrict__ stacked,
                               float* __restrict__ hx)
{
  const int i = blockIdx.x * 256 + threadIdx.x;
  hx[i] = stacked[(size_t)(T_STEPS - 1) * BATCH * HID + i];
}

// Deterministic per-call reset of the grid-sync counter (d_ws is not re-poisoned).
__global__ void init_kernel(unsigned* __restrict__ counter)
{
  if (threadIdx.x == 0 && blockIdx.x == 0)
    *counter = 0u;
}

extern "C" void kernel_launch(void* const* d_in, const int* in_sizes, int n_in,
                              void* d_out, int out_size, void* d_ws, size_t ws_size,
                              hipStream_t stream)
{
  (void)in_sizes; (void)n_in; (void)out_size; (void)ws_size;

  const float* xin = (const float*)d_in[0];
  const float* Wf  = (const float*)d_in[1];
  const float* bfv = (const float*)d_in[2];
  const float* Wi  = (const float*)d_in[3];
  const float* biv = (const float*)d_in[4];
  const float* Wg  = (const float*)d_in[5];
  const float* bgv = (const float*)d_in[6];
  const float* Wo  = (const float*)d_in[7];
  const float* bov = (const float*)d_in[8];
  const float* Wr  = (const float*)d_in[9];
  const float* br  = (const float*)d_in[10];

  // d_out layout: stacked [T,B,H] | regress [T,B,1] | hx [B,H] | cx [B,H]
  float* stacked = (float*)d_out;
  float* regress = stacked + (size_t)T_STEPS * BATCH * HID;
  float* hx      = regress + (size_t)T_STEPS * BATCH;
  float* cx      = hx + (size_t)BATCH * HID;

  // d_ws layout: [counter pad 256B][2 x B x H bf16 recurrent scratch]
  unsigned* counter = (unsigned*)d_ws;
  bf16_t*   hbuf_g  = (bf16_t*)((char*)d_ws + 256);

  init_kernel<<<dim3(1), dim3(64), 0, stream>>>(counter);
  lstm_persistent<<<dim3(NWG), dim3(256), 0, stream>>>(
      xin, Wf, bfv, Wi, biv, Wg, bgv, Wo, bov, stacked, cx, hbuf_g, counter);
  regress_kernel<<<dim3((T_STEPS * BATCH) / 8), dim3(256), 0, stream>>>(
      stacked, Wr, br, regress);
  hx_copy_kernel<<<dim3((BATCH * HID) / 256), dim3(256), 0, stream>>>(stacked, hx);
}
